// CausalSelfAttention_46145128628503
// MI455X (gfx1250) — matrix operand order repaired and checked by reading, not compiled, not run on hardware
//
#include <hip/hip_runtime.h>
#include <cstddef>
#include <cstdint>

// ---------------------------------------------------------------------------
// CDNA5 (gfx1250) bf16 WMMA causal self-attention pipeline.
// Round 3: fix async builtin signature (v4i addrspace pointers).
// ---------------------------------------------------------------------------

typedef __attribute__((ext_vector_type(16))) __bf16 bf16x16;
typedef __attribute__((ext_vector_type(8)))  __bf16 bf16x8;
typedef __attribute__((ext_vector_type(8)))  float  f32x8;
typedef __attribute__((ext_vector_type(4)))  int    i32x4;
typedef unsigned short bfraw;   // bf16 bits, host-safe parameter type

#if defined(__HIP_DEVICE_COMPILE__) && defined(__has_builtin)
#if __has_builtin(__builtin_amdgcn_global_load_async_to_lds_b128)
#define HAVE_ASYNC_LDS 1
#endif
#endif
#ifndef HAVE_ASYNC_LDS
#define HAVE_ASYNC_LDS 0
#endif

__device__ __forceinline__ bfraw f2bfu(float f) {
  unsigned u = __builtin_bit_cast(unsigned, f);
  u += 0x7fffu + ((u >> 16) & 1u);          // round-to-nearest-even
  return (bfraw)(u >> 16);
}

// Build a 16-element bf16 fragment from two contiguous 8-element (16B) chunks.
__device__ __forceinline__ bf16x16 make_frag(const bfraw* p0, const bfraw* p1) {
  union { bf16x16 v; bf16x8 h[2]; } u;
  u.h[0] = *(const bf16x8*)p0;
  u.h[1] = *(const bf16x8*)p1;
  return u.v;
}

__device__ __forceinline__ f32x8 wmma_bf16(bf16x16 a, bf16x16 b, f32x8 c) {
  // D(f32 16x16) = A(16x32 bf16) * B(32x16 bf16) + C
  return __builtin_amdgcn_wmma_f32_16x16x32_bf16(false, a, false, b,
                                                 (short)0, c, false, false);
}

// 16B per-lane async global -> LDS copy (ASYNCcnt-tracked).
__device__ __forceinline__ void async16(bfraw* lds, const bfraw* g) {
#if HAVE_ASYNC_LDS
  __builtin_amdgcn_global_load_async_to_lds_b128(
      (__attribute__((address_space(1))) i32x4*)g,
      (__attribute__((address_space(3))) i32x4*)lds, 0, 0);
#else
  (void)lds; (void)g;
#endif
}

__device__ __forceinline__ void wait_async0() {
#if HAVE_ASYNC_LDS
#if __has_builtin(__builtin_amdgcn_s_wait_asynccnt)
  __builtin_amdgcn_s_wait_asynccnt(0);
#else
  asm volatile("s_wait_asynccnt 0x0" ::: "memory");
#endif
#endif
}

__device__ __forceinline__ void wait_async4() {
#if HAVE_ASYNC_LDS
#if __has_builtin(__builtin_amdgcn_s_wait_asynccnt)
  __builtin_amdgcn_s_wait_asynccnt(4);
#else
  asm volatile("s_wait_asynccnt 0x4" ::: "memory");
#endif
#endif
}

// ---------------------------------------------------------------------------
// Stage 0: fp32 -> bf16 conversion
// ---------------------------------------------------------------------------
__global__ __launch_bounds__(256)
void cvt_f32_bf16(const float* __restrict__ src, bfraw* __restrict__ dst, int n) {
  int i = blockIdx.x * blockDim.x + threadIdx.x;
  int stride = gridDim.x * blockDim.x;
  for (; i < n; i += stride) dst[i] = f2bfu(src[i]);
}

// ---------------------------------------------------------------------------
// Stage 1: QKV GEMM.  X[16384x768] @ Wqkv[768x2304] + b.
// Block tile 128(M) x 128(N), BK=32, 8 waves (4x2), wave 32x64 (2x4 WMMA).
// Async double-buffered LDS staging.  Epilogue scatters into head-major
// Q/K/V [B, NH, S, HD] (bf16), Q scaled 1/8.
// ---------------------------------------------------------------------------
__global__ __launch_bounds__(256)
void qkv_gemm(const bfraw* __restrict__ X, const bfraw* __restrict__ W,
              const float* __restrict__ bias,
              bfraw* __restrict__ Qg, bfraw* __restrict__ Kg, bfraw* __restrict__ Vg) {
  const int K = 768, N = 2304;
  const int n0 = blockIdx.x * 128;
  const int m0 = blockIdx.y * 128;
  __shared__ bfraw Asm[2][128][32];
  __shared__ bfraw Bsm[2][32][128];
  const int tid = threadIdx.x;
  const int lane = tid & 31, w = tid >> 5;
  const int half = lane >> 4, lanelo = lane & 15;
  const int wm = w & 3, wn = w >> 2;
  const int ar = tid >> 1, ac = (tid & 1) * 16;   // A staging: 16 elems/thread
  const int br = tid >> 3, bc = (tid & 7) * 16;   // B staging: 16 elems/thread

  f32x8 acc[2][4] = {};

#if HAVE_ASYNC_LDS
  auto stage = [&](int bb, int kk) {
    const bfraw* asrc = X + (size_t)(m0 + ar) * K + kk + ac;
    async16(&Asm[bb][ar][ac],     asrc);
    async16(&Asm[bb][ar][ac + 8], asrc + 8);
    const bfraw* bsrc = W + (size_t)(kk + br) * N + n0 + bc;
    async16(&Bsm[bb][br][bc],     bsrc);
    async16(&Bsm[bb][br][bc + 8], bsrc + 8);
  };
  stage(0, 0);
  int buf = 0;
  for (int k0 = 0; k0 < K; k0 += 32) {
    __syncthreads();                 // all reads of the other buffer finished
    if (k0 + 32 < K) { stage(buf ^ 1, k0 + 32); wait_async4(); }
    else             { wait_async0(); }
    __syncthreads();                 // tile `buf` fully resident in LDS
    bf16x16 af[2];
#pragma unroll
    for (int i = 0; i < 2; ++i) {
      int m = wm * 32 + i * 16 + lanelo;
      af[i] = make_frag(&Asm[buf][m][half * 8], &Asm[buf][m][16 + half * 8]);
    }
#pragma unroll
    for (int j = 0; j < 4; ++j) {
      int n = wn * 64 + j * 16;
      bf16x16 bq;
      for (int s = 0; s < 16; ++s) bq[s] = __builtin_bit_cast(__bf16, Bsm[buf][(s & 8) * 2 + half * 8 + (s & 7)][n + lanelo]);
#pragma unroll
      for (int i = 0; i < 2; ++i) acc[i][j] = wmma_bf16(af[i], bq, acc[i][j]);
    }
    buf ^= 1;
  }
#else
  for (int k0 = 0; k0 < K; k0 += 32) {
    {
      const bfraw* asrc = X + (size_t)(m0 + ar) * K + k0 + ac;
      *(bf16x8*)&Asm[0][ar][ac]     = *(const bf16x8*)asrc;
      *(bf16x8*)&Asm[0][ar][ac + 8] = *(const bf16x8*)(asrc + 8);
      const bfraw* bsrc = W + (size_t)(k0 + br) * N + n0 + bc;
      *(bf16x8*)&Bsm[0][br][bc]     = *(const bf16x8*)bsrc;
      *(bf16x8*)&Bsm[0][br][bc + 8] = *(const bf16x8*)(bsrc + 8);
      if (k0 + 32 < K) {
        __builtin_prefetch(X + (size_t)(m0 + ar) * K + k0 + 32 + ac, 0, 3);
        __builtin_prefetch(W + (size_t)(k0 + 32 + br) * N + n0 + bc, 0, 3);
      }
    }
    __syncthreads();
    bf16x16 af[2];
#pragma unroll
    for (int i = 0; i < 2; ++i) {
      int m = wm * 32 + i * 16 + lanelo;
      af[i] = make_frag(&Asm[0][m][half * 8], &Asm[0][m][16 + half * 8]);
    }
#pragma unroll
    for (int j = 0; j < 4; ++j) {
      int n = wn * 64 + j * 16;
      bf16x16 bq;
      for (int s = 0; s < 16; ++s) bq[s] = __builtin_bit_cast(__bf16, Bsm[0][(s & 8) * 2 + half * 8 + (s & 7)][n + lanelo]);
#pragma unroll
      for (int i = 0; i < 2; ++i) acc[i][j] = wmma_bf16(af[i], bq, acc[i][j]);
    }
    __syncthreads();
  }
#endif

#pragma unroll
  for (int i = 0; i < 2; ++i)
#pragma unroll
    for (int j = 0; j < 4; ++j)
#pragma unroll
      for (int r = 0; r < 8; ++r) {
        int m = m0 + wm * 32 + i * 16 + half * 8 + r;
        int n = n0 + wn * 64 + j * 16 + lanelo;
        float v = acc[i][j][r] + bias[n];
        int which = n / 768;
        int nn = n - which * 768;
        int h = nn >> 6, d = nn & 63;
        int b = m >> 10, s = m & 1023;
        size_t idx = (((size_t)(b * 12 + h)) * 1024 + s) * 64 + d;
        if (which == 0)      Qg[idx] = f2bfu(v * 0.125f);  // fold 1/sqrt(64)
        else if (which == 1) Kg[idx] = f2bfu(v);
        else                 Vg[idx] = f2bfu(v);
      }
}

// ---------------------------------------------------------------------------
// Stage 2: causal flash attention.
// Block = one (b,h) pair x 128 q-rows; 8 waves, wave owns 16 q-rows.
// K/V streamed in tiles of 32 sequence positions; K transposed through LDS,
// V staged via async copy.
// ---------------------------------------------------------------------------
__global__ __launch_bounds__(256)
void attn_fwd(const bfraw* __restrict__ Qg, const bfraw* __restrict__ Kg,
              const bfraw* __restrict__ Vg, bfraw* __restrict__ Og) {
  const int bh = blockIdx.x;            // 0..191
  const int q0 = blockIdx.y * 128;
  const int b = bh / 12, h = bh % 12;
  const size_t base = (size_t)bh * 1024 * 64;

  __shared__ bfraw Kt[64][32];          // [d][kj]  (transposed K tile)
  __shared__ bfraw Vs[32][64];          // [kj][d]
  __shared__ bfraw Ps[8][16][32];       // per-wave P tile (D->A layout bounce)

  const int tid = threadIdx.x, lane = tid & 31, w = tid >> 5;
  const int half = lane >> 4, lanelo = lane & 15;

  // Q fragments: A-layout, loaded straight from global (contiguous 16B chunks).
  bf16x16 qf[2];
  {
    int m = q0 + w * 16 + lanelo;
    const bfraw* qrow = Qg + base + (size_t)m * 64;
#pragma unroll
    for (int t = 0; t < 2; ++t)
      qf[t] = make_frag(qrow + t * 32 + half * 8,
                        qrow + t * 32 + 16 + half * 8);
  }

  f32x8 O[4] = {};
  float rm[8], rl[8];
#pragma unroll
  for (int r = 0; r < 8; ++r) { rm[r] = -1e30f; rl[r] = 0.f; }

  const int nkt = q0 / 32 + 4;          // covers k0 <= q0+127
  for (int kt = 0; kt < nkt; ++kt) {
    const int k0 = kt * 32;
    {  // stage K^T (register transpose) and V (async) : 32 rows x 64 d each
      int sk = tid >> 3, d0 = (tid & 7) * 8;
      bf16x8 kv = *(const bf16x8*)(Kg + base + (size_t)(k0 + sk) * 64 + d0);
      union { bf16x8 v; bfraw u[8]; } ku; ku.v = kv;
#pragma unroll
      for (int u = 0; u < 8; ++u) Kt[d0 + u][sk] = ku.u[u];
      const bfraw* vp = Vg + base + (size_t)(k0 + sk) * 64 + d0;
#if HAVE_ASYNC_LDS
      async16(&Vs[sk][d0], vp);
#else
      *(bf16x8*)&Vs[sk][d0] = *(const bf16x8*)vp;
#endif
    }
    wait_async0();
    __syncthreads();

    // scores S = Q . K^T  (16 rows x 32 cols per wave)
    f32x8 s[2] = {};
#pragma unroll
    for (int t = 0; t < 2; ++t) {
      bf16x16 kf0, kf1;
      for (int s = 0; s < 16; ++s) {
        int kx = t * 32 + (s & 8) * 2 + half * 8 + (s & 7);
        kf0[s] = __builtin_bit_cast(__bf16, Kt[kx][lanelo]);
        kf1[s] = __builtin_bit_cast(__bf16, Kt[kx][16 + lanelo]);
      }
      s[0] = wmma_bf16(qf[t], kf0, s[0]);
      s[1] = wmma_bf16(qf[t], kf1, s[1]);
    }

    // causal mask
    const int rowbase = q0 + w * 16 + half * 8;
    if (k0 + 31 > rowbase) {
#pragma unroll
      for (int j = 0; j < 2; ++j) {
        int col = k0 + j * 16 + lanelo;
#pragma unroll
        for (int r = 0; r < 8; ++r)
          if (col > rowbase + r) s[j][r] = -1e30f;
      }
    }

    // online softmax (row stats live in lanes of one 16-lane half)
#pragma unroll
    for (int r = 0; r < 8; ++r) {
      float lmax = fmaxf(s[0][r], s[1][r]);
      for (int msk = 8; msk; msk >>= 1)
        lmax = fmaxf(lmax, __shfl_xor(lmax, msk, 16));
      float nm = fmaxf(rm[r], lmax);
      float sc = __expf(rm[r] - nm);
      float p0 = __expf(s[0][r] - nm);
      float p1 = __expf(s[1][r] - nm);
      float lsum = p0 + p1;
      for (int msk = 8; msk; msk >>= 1)
        lsum += __shfl_xor(lsum, msk, 16);
      rl[r] = rl[r] * sc + lsum;
      rm[r] = nm;
#pragma unroll
      for (int jo = 0; jo < 4; ++jo) O[jo][r] *= sc;
      Ps[w][half * 8 + r][lanelo]      = f2bfu(p0);
      Ps[w][half * 8 + r][16 + lanelo] = f2bfu(p1);
    }

    // O += P . V   (P re-read in A-layout from the wave's own LDS patch)
    bf16x16 pf = make_frag(&Ps[w][lanelo][half * 8],
                           &Ps[w][lanelo][16 + half * 8]);
#pragma unroll
    for (int jo = 0; jo < 4; ++jo) {
      bf16x16 vf;
      for (int s = 0; s < 16; ++s) vf[s] = __builtin_bit_cast(__bf16, Vs[(s & 8) * 2 + half * 8 + (s & 7)][jo * 16 + lanelo]);
      O[jo] = wmma_bf16(pf, vf, O[jo]);
    }
    __syncthreads();   // Kt/Vs/Ps consumed before next stage overwrites
  }

  // normalize and store [b, s, h*64+d] bf16
#pragma unroll
  for (int jo = 0; jo < 4; ++jo)
#pragma unroll
    for (int r = 0; r < 8; ++r) {
      int srow = q0 + w * 16 + half * 8 + r;
      float v = O[jo][r] / rl[r];
      Og[((size_t)(b * 1024 + srow)) * 768 + h * 64 + jo * 16 + lanelo] =
          f2bfu(v);
    }
}

// ---------------------------------------------------------------------------
// Stage 3: output projection. A[16384x768] @ Wproj[768x768] + b -> fp32 out.
// Same structure as stage 1 (BN=128, async double-buffered staging).
// ---------------------------------------------------------------------------
__global__ __launch_bounds__(256)
void proj_gemm(const bfraw* __restrict__ A, const bfraw* __restrict__ W,
               const float* __restrict__ bias, float* __restrict__ out) {
  const int K = 768, N = 768;
  const int n0 = blockIdx.x * 128;
  const int m0 = blockIdx.y * 128;
  __shared__ bfraw Asm[2][128][32];
  __shared__ bfraw Bsm[2][32][128];
  const int tid = threadIdx.x;
  const int lane = tid & 31, w = tid >> 5;
  const int half = lane >> 4, lanelo = lane & 15;
  const int wm = w & 3, wn = w >> 2;
  const int ar = tid >> 1, ac = (tid & 1) * 16;
  const int br = tid >> 3, bc = (tid & 7) * 16;

  f32x8 acc[2][4] = {};

#if HAVE_ASYNC_LDS
  auto stage = [&](int bb, int kk) {
    const bfraw* asrc = A + (size_t)(m0 + ar) * K + kk + ac;
    async16(&Asm[bb][ar][ac],     asrc);
    async16(&Asm[bb][ar][ac + 8], asrc + 8);
    const bfraw* bsrc = W + (size_t)(kk + br) * N + n0 + bc;
    async16(&Bsm[bb][br][bc],     bsrc);
    async16(&Bsm[bb][br][bc + 8], bsrc + 8);
  };
  stage(0, 0);
  int buf = 0;
  for (int k0 = 0; k0 < K; k0 += 32) {
    __syncthreads();
    if (k0 + 32 < K) { stage(buf ^ 1, k0 + 32); wait_async4(); }
    else             { wait_async0(); }
    __syncthreads();
    bf16x16 af[2];
#pragma unroll
    for (int i = 0; i < 2; ++i) {
      int m = wm * 32 + i * 16 + lanelo;
      af[i] = make_frag(&Asm[buf][m][half * 8], &Asm[buf][m][16 + half * 8]);
    }
#pragma unroll
    for (int j = 0; j < 4; ++j) {
      int n = wn * 64 + j * 16;
      bf16x16 bq;
      for (int s = 0; s < 16; ++s) bq[s] = __builtin_bit_cast(__bf16, Bsm[buf][(s & 8) * 2 + half * 8 + (s & 7)][n + lanelo]);
#pragma unroll
      for (int i = 0; i < 2; ++i) acc[i][j] = wmma_bf16(af[i], bq, acc[i][j]);
    }
    buf ^= 1;
  }
#else
  for (int k0 = 0; k0 < K; k0 += 32) {
    {
      const bfraw* asrc = A + (size_t)(m0 + ar) * K + k0 + ac;
      *(bf16x8*)&Asm[0][ar][ac]     = *(const bf16x8*)asrc;
      *(bf16x8*)&Asm[0][ar][ac + 8] = *(const bf16x8*)(asrc + 8);
      const bfraw* bsrc = W + (size_t)(k0 + br) * N + n0 + bc;
      *(bf16x8*)&Bsm[0][br][bc]     = *(const bf16x8*)bsrc;
      *(bf16x8*)&Bsm[0][br][bc + 8] = *(const bf16x8*)(bsrc + 8);
    }
    __syncthreads();
    bf16x16 af[2];
#pragma unroll
    for (int i = 0; i < 2; ++i) {
      int m = wm * 32 + i * 16 + lanelo;
      af[i] = make_frag(&Asm[0][m][half * 8], &Asm[0][m][16 + half * 8]);
    }
#pragma unroll
    for (int j = 0; j < 4; ++j) {
      int n = wn * 64 + j * 16;
      bf16x16 bq;
      for (int s = 0; s < 16; ++s) bq[s] = __builtin_bit_cast(__bf16, Bsm[0][(s & 8) * 2 + half * 8 + (s & 7)][n + lanelo]);
#pragma unroll
      for (int i = 0; i < 2; ++i) acc[i][j] = wmma_bf16(af[i], bq, acc[i][j]);
    }
    __syncthreads();
  }
#endif

#pragma unroll
  for (int i = 0; i < 2; ++i)
#pragma unroll
    for (int j = 0; j < 4; ++j)
#pragma unroll
      for (int r = 0; r < 8; ++r) {
        int m = m0 + wm * 32 + i * 16 + half * 8 + r;
        int n = n0 + wn * 64 + j * 16 + lanelo;
        out[(size_t)m * N + n] = acc[i][j][r] + bias[n];
      }
}

// ---------------------------------------------------------------------------
// Launch
// ---------------------------------------------------------------------------
extern "C" void kernel_launch(void* const* d_in, const int* in_sizes, int n_in,
                              void* d_out, int out_size, void* d_ws, size_t ws_size,
                              hipStream_t stream) {
  (void)in_sizes; (void)n_in; (void)out_size; (void)ws_size;
  const float* x     = (const float*)d_in[0];
  const float* Wqkv  = (const float*)d_in[1];
  const float* bqkv  = (const float*)d_in[2];
  const float* Wproj = (const float*)d_in[3];
  const float* bproj = (const float*)d_in[4];
  float* out = (float*)d_out;
  char* ws = (char*)d_ws;

  // workspace layout (bytes, 256-aligned)
  bfraw* xb     = (bfraw*)(ws + 0);              // 16*1024*768 bf16 = 25165824 B
  bfraw* wqkvb  = (bfraw*)(ws + 25165824);       // 768*2304    bf16 =  3538944 B
  bfraw* wprojb = (bfraw*)(ws + 28704768);       // 768*768     bf16 =  1179648 B
  bfraw* Qg     = (bfraw*)(ws + 29884416);       // head-major Q
  bfraw* Kg     = (bfraw*)(ws + 55050240);       // head-major K
  bfraw* Vg     = (bfraw*)(ws + 80216064);       // head-major V
  bfraw* Ab     = (bfraw*)(ws + 105381888);      // attention output [B,S,H]

  const int NX = 16 * 1024 * 768;
  const int NWQ = 768 * 2304;
  const int NWP = 768 * 768;

  cvt_f32_bf16<<<4096, 256, 0, stream>>>(x,     xb,     NX);
  cvt_f32_bf16<<<2048, 256, 0, stream>>>(Wqkv,  wqkvb,  NWQ);
  cvt_f32_bf16<<<1024, 256, 0, stream>>>(Wproj, wprojb, NWP);

  qkv_gemm<<<dim3(2304 / 128, 16384 / 128), 256, 0, stream>>>(
      xb, wqkvb, bqkv, Qg, Kg, Vg);

  attn_fwd<<<dim3(16 * 12, 1024 / 128), 256, 0, stream>>>(Qg, Kg, Vg, Ab);

  proj_gemm<<<dim3(768 / 128, 16384 / 128), 256, 0, stream>>>(
      Ab, wprojb, bproj, out);
}
